// P2PDecoderLayer_62259845923550
// MI455X (gfx1250) — matrix operand order repaired and checked by reading, not compiled, not run on hardware
//
#include <hip/hip_runtime.h>

typedef _Float16 f16;
typedef __attribute__((ext_vector_type(16))) _Float16 v16h;
typedef __attribute__((ext_vector_type(8)))  _Float16 v8h;
typedef __attribute__((ext_vector_type(8)))  float    v8f;

#define DM   512
#define NH   8
#define HDIM 64
#define BB   2
#define FFD  2048

// ============================================================================
// Batched WMMA GEMM with CDNA5 async global->LDS double-buffered pipeline.
//   C[z] = scale * (A[z] x B[z]^T) + bias, optional ReLU.
// A: (M,K) rows at A + zb*a_b + zh*a_h + m*a_rs   (K contiguous, f16)
// B: (N,K) rows at Bm + zb*b_b + zh*b_h + n*b_rs  (K contiguous, f16)
// C: f32 at C + zb*c_b + zh*c_h + m*c_rs + n ;  z=blockIdx.z, zb=z/Hdiv, zh=z%Hdiv
// Block = 256 threads = 8 waves; tile BM=128 x BN (64 or 128), BK=32.
// Wave grid 4(M) x 2(N): each wave computes 32 x (BN/2) of C.
// Tiles stream into LDS with GLOBAL_LOAD_ASYNC_TO_LDS_B128 (ASYNCcnt),
// double buffered; WMMA consumes from LDS (ds_load_b128 fragments).
// Requires M%128==0, N%BN==0, K%32==0 (all shapes in this layer satisfy this).
// ============================================================================
template <int BN>
__global__ __launch_bounds__(256) void wmma_gemm_lds(
    const f16* __restrict__ A, long long a_rs, long long a_b, long long a_h,
    const f16* __restrict__ Bm, long long b_rs, long long b_b, long long b_h,
    float* __restrict__ C, long long c_rs, long long c_b, long long c_h,
    const float* __restrict__ bias, int M, int N, int K, int Hdiv,
    float scale, int relu)
{
  constexpr int NF = BN / 32;            // N fragments per wave
  __shared__ f16 As[2][128 * 32];
  __shared__ f16 Bs[2][BN * 32];

  int z  = blockIdx.z;
  int zb = z / Hdiv, zh = z % Hdiv;
  A  += zb * a_b + zh * a_h;
  Bm += zb * b_b + zh * b_h;
  C  += zb * c_b + zh * c_h;

  int tid  = threadIdx.x;
  int wid  = tid >> 5;
  int lane = tid & 31;
  int r  = lane & 15;                    // row within fragment
  int hi = lane >> 4;                    // lane-half selector
  int wm = wid & 3;                      // wave row-block (4 x 32 rows)
  int wn = wid >> 2;                     // wave col-block (2 x BN/2 cols)
  int m0 = blockIdx.y * 128;
  int n0 = blockIdx.x * BN;

  v8f acc[2][NF];
  v8f zero = {};
#pragma unroll
  for (int mf = 0; mf < 2; ++mf)
#pragma unroll
    for (int nf = 0; nf < NF; ++nf) acc[mf][nf] = zero;

  // ---- async global->LDS staging of one 32-wide K slab into buffer `buf` ----
  auto stage = [&](int buf, int k0) {
#pragma unroll
    for (int i = 0; i < 2; ++i) {                 // A: 128 rows x 4 chunks / 256 thr
      int c = tid + i * 256;
      int row = c >> 2, off = (c & 3) * 8;        // 8-half (16B) chunks
      const f16* g = A + (long long)(m0 + row) * a_rs + k0 + off;
      unsigned l = (unsigned)(size_t)(const void*)&As[buf][row * 32 + off];
      asm volatile("global_load_async_to_lds_b128 %0, %1, off"
                   :: "v"(l), "v"(g) : "memory");
    }
#pragma unroll
    for (int i = 0; i < (BN * 4) / 256; ++i) {    // B: BN rows x 4 chunks / 256 thr
      int c = tid + i * 256;
      int row = c >> 2, off = (c & 3) * 8;
      const f16* g = Bm + (long long)(n0 + row) * b_rs + k0 + off;
      unsigned l = (unsigned)(size_t)(const void*)&Bs[buf][row * 32 + off];
      asm volatile("global_load_async_to_lds_b128 %0, %1, off"
                   :: "v"(l), "v"(g) : "memory");
    }
  };

  // ---- WMMA over one staged K slab ----
  auto compute = [&](int buf) {
#pragma unroll
    for (int mf = 0; mf < 2; ++mf) {
      // A fragment (16x32): lane r = M row; chunks at K = hi*8 and 16+hi*8
      const f16* al = &As[buf][(wm * 32 + mf * 16 + r) * 32 + hi * 8];
      v8h a0 = *(const v8h*)(al);
      v8h a1 = *(const v8h*)(al + 16);
      v16h af = __builtin_shufflevector(a0, a1, 0,1,2,3,4,5,6,7,8,9,10,11,12,13,14,15);
#pragma unroll
      for (int nf = 0; nf < NF; ++nf) {
        // B fragment (32x16): lane r = N col; chunks at K = hi*8 and 16+hi*8
        const f16* bl = &Bs[buf][(wn * (BN / 2) + nf * 16 + r) * 32 + hi * 8];
        v8h b0 = *(const v8h*)(bl);
        v8h b1 = *(const v8h*)(bl + 16);
        v16h bf = __builtin_shufflevector(b0, b1, 0,1,2,3,4,5,6,7,8,9,10,11,12,13,14,15);
        acc[mf][nf] = __builtin_amdgcn_wmma_f32_16x16x32_f16(
            false, af, false, bf, (short)0, acc[mf][nf], false, false);
      }
    }
  };

  // ---- double-buffered pipeline ----
  stage(0, 0);
  asm volatile("s_wait_asynccnt 0" ::: "memory");
  __syncthreads();
  int cur = 0;
  for (int k0 = 32; k0 < K; k0 += 32) {
    stage(cur ^ 1, k0);          // stream next slab while computing current
    compute(cur);
    asm volatile("s_wait_asynccnt 0" ::: "memory");
    __syncthreads();
    cur ^= 1;
  }
  compute(cur);

  // ---- epilogue: C/D layout — VGPR v, lane l -> row +hi*8+v, col +.. + (l&15) ----
#pragma unroll
  for (int mf = 0; mf < 2; ++mf)
#pragma unroll
    for (int nf = 0; nf < NF; ++nf) {
      int n = n0 + wn * (BN / 2) + nf * 16 + r;
      float bv = bias ? bias[n] : 0.f;
#pragma unroll
      for (int v = 0; v < 8; ++v) {
        int m = m0 + wm * 32 + mf * 16 + hi * 8 + v;
        float val = acc[mf][nf][v] * scale + bv;
        if (relu) val = fmaxf(val, 0.f);
        C[(long long)m * c_rs + n] = val;
      }
    }
}

// ---------------------------------------------------------------------------
__global__ __launch_bounds__(256) void f32_to_f16_k(const float* __restrict__ s,
                                                    f16* __restrict__ d, long long n) {
  long long i = (long long)blockIdx.x * 256 + threadIdx.x;
  if (i < n) d[i] = (f16)s[i];
}

__global__ __launch_bounds__(256) void copy_f32_k(const float* __restrict__ s,
                                                  float* __restrict__ d, long long n) {
  long long i = (long long)blockIdx.x * 256 + threadIdx.x;
  if (i < n) d[i] = s[i];
}

// In-place row softmax, one block per row of length S.
__global__ __launch_bounds__(256) void softmax_rows_k(float* __restrict__ x, int S) {
  __shared__ float red[256];
  float* p = x + (long long)blockIdx.x * S;
  int t = threadIdx.x;
  float m = -3.4e38f;
  for (int i = t; i < S; i += 256) m = fmaxf(m, p[i]);
  red[t] = m; __syncthreads();
  for (int o = 128; o > 0; o >>= 1) { if (t < o) red[t] = fmaxf(red[t], red[t + o]); __syncthreads(); }
  m = red[0]; __syncthreads();
  float sum = 0.f;
  for (int i = t; i < S; i += 256) { float e = expf(p[i] - m); p[i] = e; sum += e; }
  red[t] = sum; __syncthreads();
  for (int o = 128; o > 0; o >>= 1) { if (t < o) red[t] += red[t + o]; __syncthreads(); }
  float inv = 1.f / red[0];
  for (int i = t; i < S; i += 256) p[i] *= inv;
}

// w[b,l,s] = mean_h P[b,h,l,s]
__global__ __launch_bounds__(256) void head_mean_k(const float* __restrict__ P,
                                                   float* __restrict__ w, int Lq, int Skv) {
  long long i = (long long)blockIdx.x * 256 + threadIdx.x;
  long long total = (long long)BB * Lq * Skv;
  if (i >= total) return;
  long long s = i % Skv; long long rem = i / Skv;
  long long l = rem % Lq; long long b = rem / Lq;
  long long LS = (long long)Lq * Skv;
  const float* base = P + (b * NH) * LS + l * (long long)Skv + s;
  float acc = 0.f;
#pragma unroll
  for (int h = 0; h < NH; ++h) acc += base[(long long)h * LS];
  w[i] = acc * (1.f / NH);
}

// Vt[((b*NH+h)*HDIM + n)*Skv + s] = src[(s*BB+b)*ld + coloff + h*HDIM + n]  (f32 -> f16)
__global__ __launch_bounds__(256) void pack_vT_k(const float* __restrict__ src, int ld, int coloff,
                                                 f16* __restrict__ dst, int Skv) {
  long long i = (long long)blockIdx.x * 256 + threadIdx.x;
  long long total = (long long)BB * NH * HDIM * Skv;
  if (i >= total) return;
  int s = (int)(i % Skv); long long rem = i / Skv;
  int n = (int)(rem % HDIM); int z = (int)(rem / HDIM);
  int b = z / NH, h = z % NH;
  dst[i] = (f16)src[((long long)s * BB + b) * ld + coloff + h * HDIM + n];
}

__global__ __launch_bounds__(256) void fill_i32_k(int* __restrict__ p, int val, int n) {
  int i = blockIdx.x * 256 + threadIdx.x;
  if (i < n) p[i] = val;
}

__global__ __launch_bounds__(256) void scatter_map_k(int* __restrict__ map,
                                                     const int* __restrict__ ind, int n) {
  int i = blockIdx.x * 256 + threadIdx.x;
  if (i < n) map[ind[i]] = i;
}

// dst[(s*BB+b)*DM + c] (f16) = pair-assembled row: sp if ind[s] was scattered, else entire_pair
__global__ __launch_bounds__(256) void gather_pair_k(const float* __restrict__ entire,
                                                     const float* __restrict__ sp,
                                                     const int* __restrict__ ind,
                                                     const int* __restrict__ map,
                                                     f16* __restrict__ dst, int Ssel) {
  long long i = (long long)blockIdx.x * 256 + threadIdx.x;
  long long total = (long long)Ssel * BB * DM;
  if (i >= total) return;
  int c = (int)(i % DM); long long rem = i / DM;
  int b = (int)(rem % BB); int s = (int)(rem / BB);
  int idx = ind[s];
  int j = map[idx];
  float v = (j >= 0) ? sp[((long long)j * BB + b) * DM + c]
                     : entire[((long long)idx * BB + b) * DM + c];
  dst[i] = (f16)v;
}

// y = LayerNorm(x + u1 [+ u2]) * g + be   (rows of 512; 256 threads, 2 elems each)
__global__ __launch_bounds__(256) void resid_ln_k(
    const float* __restrict__ x, const float* __restrict__ u1, const float* __restrict__ u2,
    const float* __restrict__ g, const float* __restrict__ be,
    float* __restrict__ y, f16* __restrict__ yh) {
  __shared__ float s1[256], s2[256];
  long long row = blockIdx.x;
  const float* xp  = x  + row * DM;
  const float* u1p = u1 + row * DM;
  const float* u2p = u2 ? (u2 + row * DM) : nullptr;
  int t = threadIdx.x;
  float a = xp[t]       + u1p[t]       + (u2p ? u2p[t]       : 0.f);
  float b = xp[t + 256] + u1p[t + 256] + (u2p ? u2p[t + 256] : 0.f);
  s1[t] = a + b; s2[t] = a * a + b * b;
  __syncthreads();
  for (int o = 128; o > 0; o >>= 1) {
    if (t < o) { s1[t] += s1[t + o]; s2[t] += s2[t + o]; }
    __syncthreads();
  }
  float mean = s1[0] * (1.f / DM);
  float var  = s2[0] * (1.f / DM) - mean * mean;
  float rstd = rsqrtf(var + 1e-5f);
  float o1 = (a - mean) * rstd * g[t]       + be[t];
  float o2 = (b - mean) * rstd * g[t + 256] + be[t + 256];
  float* yp = y + row * DM;
  yp[t] = o1; yp[t + 256] = o2;
  if (yh) { f16* hp = yh + row * DM; hp[t] = (f16)o1; hp[t + 256] = (f16)o2; }
}

// ============================================================================
extern "C" void kernel_launch(void* const* d_in, const int* in_sizes, int n_in,
                              void* d_out, int out_size, void* d_ws, size_t ws_size,
                              hipStream_t stream) {
  (void)in_sizes; (void)n_in; (void)out_size; (void)ws_size;

  const int Lp = 1024, Lu = 256, Ssel = 2048;

  // ---- inputs (setup_inputs dict order) ----
  const float* x_pair  = (const float*)d_in[0];
  const float* x_unary = (const float*)d_in[1];
  const float* entire  = (const float*)d_in[2];
  const int*   ind_pair = (const int*)d_in[3];
  const int*   ind_e2e  = (const int*)d_in[4];
  const int*   ind_n2e  = (const int*)d_in[5];
  const float *Wi_f[5], *bi_f[5], *Wo_f[5], *bo_f[5];
  for (int i = 0; i < 5; ++i) {
    Wi_f[i] = (const float*)d_in[6 + 4 * i];
    bi_f[i] = (const float*)d_in[7 + 4 * i];
    Wo_f[i] = (const float*)d_in[8 + 4 * i];
    bo_f[i] = (const float*)d_in[9 + 4 * i];
  }
  const float* W1   = (const float*)d_in[26]; const float* b1f  = (const float*)d_in[27];
  const float* W2   = (const float*)d_in[28]; const float* b2f  = (const float*)d_in[29];
  const float* W1u  = (const float*)d_in[30]; const float* b1fu = (const float*)d_in[31];
  const float* W2u  = (const float*)d_in[32]; const float* b2fu = (const float*)d_in[33];
  const float* g1  = (const float*)d_in[34]; const float* g1u = (const float*)d_in[35];
  const float* g2  = (const float*)d_in[36]; const float* g3  = (const float*)d_in[37];
  const float* be1 = (const float*)d_in[38]; const float* be1u = (const float*)d_in[39];
  const float* be2 = (const float*)d_in[40]; const float* be3  = (const float*)d_in[41];

  // ---- output layout (return order, f32) ----
  float* out_f = (float*)d_out;
  const long long n_uu = (long long)Lu * BB * DM;
  const long long n_up = (long long)Lp * BB * DM;
  const long long n_we = (long long)BB * Lp * Ssel;
  const long long n_wn2e = (long long)BB * Lu * Ssel;
  float* o_uu   = out_f;
  float* o_up   = o_uu + n_uu;
  float* o_we_a = o_up + n_up;
  float* o_we_b = o_we_a + n_we;
  float* o_wn2e = o_we_b + n_we;
  float* o_wn2n = o_wn2e + n_wn2e;

  // ---- workspace bump allocator (256B aligned) ----
  char* wp = (char*)d_ws;
  auto bump = [&](long long bytes) -> void* {
    void* r = wp; wp += (bytes + 255) & ~255LL; return r;
  };
  f16* Wi_h[5]; f16* Wo_h[5];
  for (int i = 0; i < 5; ++i) {
    Wi_h[i] = (f16*)bump(3LL * DM * DM * 2);
    Wo_h[i] = (f16*)bump((long long)DM * DM * 2);
  }
  f16* w1_h  = (f16*)bump((long long)FFD * DM * 2);
  f16* w2_h  = (f16*)bump((long long)DM * FFD * 2);
  f16* w1u_h = (f16*)bump((long long)FFD * DM * 2);
  f16* w2u_h = (f16*)bump((long long)DM * FFD * 2);
  f16* xp_h  = (f16*)bump((long long)Lp * BB * DM * 2);
  f16* xu_h  = (f16*)bump((long long)Lu * BB * DM * 2);
  float* sp  = (float*)bump((long long)Lp * BB * DM * 4);
  f16*   sp_h = (f16*)bump((long long)Lp * BB * DM * 2);
  float* su  = (float*)bump((long long)Lu * BB * DM * 4);
  f16*   su_h = (f16*)bump((long long)Lu * BB * DM * 2);
  f16* pe2e_h = (f16*)bump((long long)Ssel * BB * DM * 2);
  f16* pn2e_h = (f16*)bump((long long)Ssel * BB * DM * 2);
  int* map = (int*)bump(4096 * 4);
  float* up_e2e = (float*)bump((long long)Lp * BB * DM * 4);
  float* up_n2e = (float*)bump((long long)Lu * BB * DM * 4);
  float* up_n2n = (float*)bump((long long)Lu * BB * DM * 4);
  float* attn_tmp = (float*)bump((long long)Lp * BB * DM * 4);
  float* t   = (float*)bump((long long)Lp * BB * DM * 4);
  f16*   t_h = (f16*)bump((long long)Lp * BB * DM * 2);
  float* tu  = (float*)bump((long long)Lu * BB * DM * 4);
  f16*   tu_h = (f16*)bump((long long)Lu * BB * DM * 2);
  float* h1   = (float*)bump((long long)Lp * BB * FFD * 4);
  f16*   h1_h = (f16*)bump((long long)Lp * BB * FFD * 2);
  float* h2   = (float*)bump((long long)Lp * BB * DM * 4);
  float* qb   = (float*)bump((long long)Lp * BB * DM * 4);
  f16*   qb_h = (f16*)bump((long long)Lp * BB * DM * 2);
  float* kvb   = (float*)bump((long long)Ssel * BB * 2 * DM * 4);
  f16*   kvb_h = (f16*)bump((long long)Ssel * BB * 2 * DM * 2);
  float* Pbuf = (float*)bump((long long)BB * NH * Lp * Ssel * 4);
  f16*   P_h  = (f16*)bump((long long)BB * NH * Lp * Ssel * 2);
  f16*   vt_h = (f16*)bump((long long)BB * NH * HDIM * Ssel * 2);
  float* ctx   = (float*)bump((long long)Lp * BB * DM * 4);
  f16*   ctx_h = (f16*)bump((long long)Lp * BB * DM * 2);

  // ---- launch helpers ----
  auto cvt = [&](const float* s, f16* d, long long n) {
    f32_to_f16_k<<<dim3((unsigned)((n + 255) / 256)), dim3(256), 0, stream>>>(s, d, n);
  };
  auto gemm = [&](const f16* A, long long ars, long long ab, long long ah,
                  const f16* Bm, long long brs, long long bb2, long long bh,
                  float* C, long long crs, long long cb, long long ch,
                  const float* bias, int M, int N, int K, int Hdiv, int Z,
                  float scale, int relu) {
    if (N % 128 == 0) {
      dim3 g((unsigned)(N / 128), (unsigned)(M / 128), (unsigned)Z);
      wmma_gemm_lds<128><<<g, dim3(256), 0, stream>>>(A, ars, ab, ah, Bm, brs, bb2, bh,
                                                      C, crs, cb, ch, bias, M, N, K, Hdiv, scale, relu);
    } else {
      dim3 g((unsigned)(N / 64), (unsigned)(M / 128), (unsigned)Z);
      wmma_gemm_lds<64><<<g, dim3(256), 0, stream>>>(A, ars, ab, ah, Bm, brs, bb2, bh,
                                                     C, crs, cb, ch, bias, M, N, K, Hdiv, scale, relu);
    }
  };

  // Full MHA block: out (Lq*BB, DM) f32; optional averaged+re-softmaxed weights to wdst.
  auto run_mha = [&](const f16* qin, int Lq, const f16* kvin, int Skv,
                     const f16* Wih, const float* bi, const f16* Woh, const float* bo,
                     float* out, float* wdst) {
    int Mq = Lq * BB, Mkv = Skv * BB;
    long long LS = (long long)Lq * Skv;
    // Q projection (Wi rows 0..511)
    gemm(qin, DM, 0, 0, Wih, DM, 0, 0, qb, DM, 0, 0, bi, Mq, DM, DM, 1, 1, 1.f, 0);
    cvt(qb, qb_h, (long long)Mq * DM);
    // K,V projection (Wi rows 512..1535) -> kvb cols [0:512]=K, [512:1024]=V
    gemm(kvin, DM, 0, 0, Wih + (long long)DM * DM, DM, 0, 0, kvb, 2 * DM, 0, 0,
         bi + DM, Mkv, 2 * DM, DM, 1, 1, 1.f, 0);
    cvt(kvb, kvb_h, (long long)Mkv * 2 * DM);
    // Logits: P[b,h,l,s] = (1/8) q_h . k_h   (batched over z=b*8+h)
    gemm(qb_h, (long long)BB * DM, DM, HDIM,
         kvb_h, (long long)BB * 2 * DM, 2 * DM, HDIM,
         Pbuf, Skv, (long long)NH * LS, LS,
         nullptr, Lq, Skv, HDIM, NH, BB * NH, 0.125f, 0);
    softmax_rows_k<<<dim3((unsigned)(BB * NH * Lq)), dim3(256), 0, stream>>>(Pbuf, Skv);
    if (wdst) {
      long long tot = (long long)BB * Lq * Skv;
      head_mean_k<<<dim3((unsigned)((tot + 255) / 256)), dim3(256), 0, stream>>>(Pbuf, wdst, Lq, Skv);
      softmax_rows_k<<<dim3((unsigned)(BB * Lq)), dim3(256), 0, stream>>>(wdst, Skv);  // ref double-softmax
    }
    cvt(Pbuf, P_h, (long long)BB * NH * LS);
    long long vtn = (long long)BB * NH * HDIM * Skv;
    pack_vT_k<<<dim3((unsigned)((vtn + 255) / 256)), dim3(256), 0, stream>>>(kvb, 2 * DM, DM, vt_h, Skv);
    // ctx[b,h,l,:] = P x V   -> written interleaved as (l*BB+b, h*64+n)   (BN=64 path)
    gemm(P_h, Skv, (long long)NH * LS, LS,
         vt_h, Skv, (long long)NH * HDIM * Skv, (long long)HDIM * Skv,
         ctx, (long long)BB * DM, DM, HDIM,
         nullptr, Lq, HDIM, Skv, NH, BB * NH, 1.f, 0);
    cvt(ctx, ctx_h, (long long)Mq * DM);
    // Output projection
    gemm(ctx_h, DM, 0, 0, Woh, DM, 0, 0, out, DM, 0, 0, bo, Mq, DM, DM, 1, 1, 1.f, 0);
  };

  // ---- weight conversion (f32 -> f16) ----
  for (int i = 0; i < 5; ++i) {
    cvt(Wi_f[i], Wi_h[i], 3LL * DM * DM);
    cvt(Wo_f[i], Wo_h[i], (long long)DM * DM);
  }
  cvt(W1, w1_h, (long long)FFD * DM);  cvt(W2, w2_h, (long long)DM * FFD);
  cvt(W1u, w1u_h, (long long)FFD * DM); cvt(W2u, w2u_h, (long long)DM * FFD);
  cvt(x_pair, xp_h, (long long)Lp * BB * DM);
  cvt(x_unary, xu_h, (long long)Lu * BB * DM);

  // ---- self-attention blocks + post-norm ----
  run_mha(xp_h, Lp, xp_h, Lp, Wi_h[0], bi_f[0], Wo_h[0], bo_f[0], attn_tmp, nullptr);
  resid_ln_k<<<dim3((unsigned)(Lp * BB)), dim3(256), 0, stream>>>(x_pair, attn_tmp, nullptr, g1, be1, sp, sp_h);
  run_mha(xu_h, Lu, xu_h, Lu, Wi_h[1], bi_f[1], Wo_h[1], bo_f[1], attn_tmp, nullptr);
  resid_ln_k<<<dim3((unsigned)(Lu * BB)), dim3(256), 0, stream>>>(x_unary, attn_tmp, nullptr, g1u, be1u, su, su_h);

  // ---- pair assembly: pair = entire_pair with rows ind_pair replaced by sp ----
  fill_i32_k<<<dim3(16), dim3(256), 0, stream>>>(map, -1, 4096);
  scatter_map_k<<<dim3(4), dim3(256), 0, stream>>>(map, ind_pair, 1024);
  {
    long long tot = (long long)Ssel * BB * DM;
    gather_pair_k<<<dim3((unsigned)((tot + 255) / 256)), dim3(256), 0, stream>>>(entire, sp, ind_e2e, map, pe2e_h, Ssel);
    gather_pair_k<<<dim3((unsigned)((tot + 255) / 256)), dim3(256), 0, stream>>>(entire, sp, ind_n2e, map, pn2e_h, Ssel);
  }

  // ---- cross-attention blocks (with averaged-weight outputs) ----
  run_mha(sp_h, Lp, pe2e_h, Ssel, Wi_h[2], bi_f[2], Wo_h[2], bo_f[2], up_e2e, o_we_a);
  run_mha(su_h, Lu, pn2e_h, Ssel, Wi_h[3], bi_f[3], Wo_h[3], bo_f[3], up_n2e, o_wn2e);
  run_mha(su_h, Lu, su_h, Lu,     Wi_h[4], bi_f[4], Wo_h[4], bo_f[4], up_n2n, o_wn2n);
  copy_f32_k<<<dim3((unsigned)((n_we + 255) / 256)), dim3(256), 0, stream>>>(o_we_a, o_we_b, n_we);

  // ---- pair path: LN(sp + 2*up_e2e), FFN, LN ----
  resid_ln_k<<<dim3((unsigned)(Lp * BB)), dim3(256), 0, stream>>>(sp, up_e2e, up_e2e, g2, be2, t, t_h);
  gemm(t_h, DM, 0, 0, w1_h, DM, 0, 0, h1, FFD, 0, 0, b1f, Lp * BB, FFD, DM, 1, 1, 1.f, 1);
  cvt(h1, h1_h, (long long)Lp * BB * FFD);
  gemm(h1_h, FFD, 0, 0, w2_h, FFD, 0, 0, h2, DM, 0, 0, b2f, Lp * BB, DM, FFD, 1, 1, 1.f, 0);
  resid_ln_k<<<dim3((unsigned)(Lp * BB)), dim3(256), 0, stream>>>(t, h2, nullptr, g3, be3, o_up, nullptr);

  // ---- unary path: LN(su + up_n2e + up_n2n), FFN, LN ----
  resid_ln_k<<<dim3((unsigned)(Lu * BB)), dim3(256), 0, stream>>>(su, up_n2e, up_n2n, g2, be2, tu, tu_h);
  gemm(tu_h, DM, 0, 0, w1u_h, DM, 0, 0, h1, FFD, 0, 0, b1fu, Lu * BB, FFD, DM, 1, 1, 1.f, 1);
  cvt(h1, h1_h, (long long)Lu * BB * FFD);
  gemm(h1_h, FFD, 0, 0, w2u_h, FFD, 0, 0, h2, DM, 0, 0, b2fu, Lu * BB, DM, FFD, 1, 1, 1.f, 0);
  resid_ln_k<<<dim3((unsigned)(Lu * BB)), dim3(256), 0, stream>>>(tu, h2, nullptr, g3, be3, o_uu, nullptr);
}
